// FirstAttentiveTransformer_4406636446338
// MI455X (gfx1250) — matrix operand order repaired and checked by reading, not compiled, not run on hardware
//
#include <hip/hip_runtime.h>
#include <hip/hip_bf16.h>
#include <stdint.h>

// MI455X / gfx1250: wave32, WMMA 16x16x32 bf16 (f32 acc), TDM tensor_load_to_lds
// double-buffered pipeline. Split-bf16 (bf16x3) GEMMs for near-fp32 accuracy.

typedef unsigned short u16;
typedef __bf16 bf16x16 __attribute__((ext_vector_type(16)));
typedef float  f32x8   __attribute__((ext_vector_type(8)));
typedef unsigned int u32x4 __attribute__((ext_vector_type(4)));
typedef int i32x4 __attribute__((ext_vector_type(4)));
typedef int i32x8 __attribute__((ext_vector_type(8)));

static constexpr int Bsz = 16384;
static constexpr int Dd  = 512;

#if defined(__has_builtin) && defined(__AMDGCN__)
#if __has_builtin(__builtin_amdgcn_tensor_load_to_lds)
#define HAVE_TDM 1
#endif
#endif
#ifndef HAVE_TDM
#define HAVE_TDM 0
#endif

// ---------------- bf16 split helpers ----------------
__device__ __forceinline__ u16 f2bf_rne(float f) {
  unsigned u = __float_as_uint(f);
  u += 0x7fffu + ((u >> 16) & 1u);
  return (u16)(u >> 16);
}
__device__ __forceinline__ float bf2f(u16 h) {
  return __uint_as_float(((unsigned)h) << 16);
}
__device__ __forceinline__ void split2(float f, u16 &hi, u16 &lo) {
  hi = f2bf_rne(f);
  lo = f2bf_rne(f - bf2f(hi));
}

// ---------------- TDM: 2D tile load, 128 rows x 64B, LDS row stride 80B ----------------
// D# per CDNA5 ISA ch.8: group0 {count, lds_addr, global_addr, type=2};
// group1 {data_size=4B, pad_enable, pad_interval=64B, pad_amount=16B,
//         tensor_dim0=16 dw, tensor_dim1=128, tile 16x128, dim0_stride}
#if HAVE_TDM
__device__ __forceinline__ void tdm_load_tile(const u16* gsrc, unsigned lds_byte_addr,
                                              unsigned row_stride_dwords) {
  unsigned long long ga = (unsigned long long)(size_t)gsrc;
  u32x4 g0;
  g0[0] = 1u;                                                 // count=1 (user mode)
  g0[1] = lds_byte_addr;                                      // lds_addr
  g0[2] = (unsigned)(ga & 0xffffffffu);                       // global_addr[31:0]
  g0[3] = (unsigned)((ga >> 32) & 0x01ffffffu) | (2u << 30);  // global_addr[56:32] | type=2
  i32x8 g1;
  const unsigned td0 = 16, td1 = 128, tile0 = 16, tile1 = 128;
  // word0: wg_mask=0 | data_size(2=4B)<<16 | pad_en<<20 | pad_interval(3:64B)<<22 | pad_amount(3:16B)<<25
  g1[0] = (int)((2u << 16) | (1u << 20) | (3u << 22) | (3u << 25));
  g1[1] = (int)((td0 & 0xffffu) << 16);                         // tensor_dim0[15:0] @ [63:48]
  g1[2] = (int)(((td0 >> 16) & 0xffffu) | ((td1 & 0xffffu) << 16)); // dim0[31:16] | dim1[15:0]
  g1[3] = (int)(((td1 >> 16) & 0xffffu) | (tile0 << 16));       // dim1[31:16] | tile_dim0
  g1[4] = (int)(tile1 | (0u << 16));                            // tile_dim1 | tile_dim2=0
  g1[5] = (int)row_stride_dwords;                               // tensor_dim0_stride[31:0]
  g1[6] = 0;                                                    // stride[47:32] | dim1_stride[15:0]
  g1[7] = 0;
  i32x4 z4 = (i32x4)0;
#if __clang_major__ >= 23
  i32x8 z8 = (i32x8)0;
  __builtin_amdgcn_tensor_load_to_lds(g0, g1, z4, z4, z8, 0);
#else
  __builtin_amdgcn_tensor_load_to_lds(g0, g1, z4, z4, 0);
#endif
}
#endif

// ---------------- prep kernels ----------------
__global__ void prep_x_kernel(const float* __restrict__ x,
                              const float* __restrict__ gamma, const float* __restrict__ beta,
                              const float* __restrict__ mean,  const float* __restrict__ var,
                              u16* __restrict__ Xhi, u16* __restrict__ Xlo,
                              u16* __restrict__ Phi, u16* __restrict__ Plo) {
  int idx = blockIdx.x * 256 + threadIdx.x;
  int d = idx & (Dd - 1);
  int b = idx >> 9;
  float xv = x[idx];
  u16 h, l;
  split2(xv, h, l); Xhi[idx] = h; Xlo[idx] = l;
  float xb = (xv - mean[d]) * rsqrtf(var[d] + 1e-3f) * gamma[d] + beta[d];
  size_t p = (size_t)b * 1024 + d;
  split2(xb, h, l);      Phi[p] = h;       Plo[p] = l;
  split2(xb * xb, h, l); Phi[p + 512] = h; Plo[p + 512] = l;
}

// f32 [K,N] -> transposed split bf16 [N,K]
__global__ void split_wt_kernel(const float* __restrict__ src, u16* __restrict__ hi,
                                u16* __restrict__ lo, int K, int N) {
  int idx = blockIdx.x * 256 + threadIdx.x;
  if (idx >= K * N) return;
  int k = idx / N, n = idx - k * N;
  u16 h, l; split2(src[idx], h, l);
  size_t o = (size_t)n * K + k;
  hi[o] = h; lo[o] = l;
}

// Wpt[j, d]       = -2*coefs[j,d]^2*centers[j,d]   (multiplies xb)
// Wpt[j, 512+d]   =  coefs[j,d]^2                  (multiplies xb^2)   -- [N=512, K=1024]
__global__ void prep_wp_kernel(const float* __restrict__ centers, const float* __restrict__ coefs,
                               u16* __restrict__ Wphi, u16* __restrict__ Wplo) {
  int idx = blockIdx.x * 256 + threadIdx.x;   // over 512*512
  int j = idx >> 9, d = idx & 511;
  float c  = coefs[(size_t)j * Dd + d];
  float ce = centers[(size_t)j * Dd + d];
  float a  = c * c;
  u16 h, l;
  split2(-2.0f * a * ce, h, l);
  Wphi[(size_t)j * 1024 + d] = h;        Wplo[(size_t)j * 1024 + d] = l;
  split2(a, h, l);
  Wphi[(size_t)j * 1024 + 512 + d] = h;  Wplo[(size_t)j * 1024 + 512 + d] = l;
}

__global__ void prep_cvec_kernel(const float* __restrict__ centers, const float* __restrict__ coefs,
                                 float* __restrict__ cvec) {
  __shared__ float red[256];
  int j = blockIdx.x, t = threadIdx.x;
  float s = 0.f;
  for (int d = t; d < Dd; d += 256) {
    float c  = coefs[(size_t)j * Dd + d];
    float ce = centers[(size_t)j * Dd + d];
    s += c * c * ce * ce;
  }
  red[t] = s; __syncthreads();
  for (int off = 128; off > 0; off >>= 1) { if (t < off) red[t] += red[t + off]; __syncthreads(); }
  if (t == 0) cvec[j] = red[0];
}

// ---------------- WMMA inner step (shared by TDM / fallback paths) ----------------
// A lane<16: K 0..7 & 16..23 ; lane>=16: K 8..15 & 24..31 (M = lane%16)
// B (transposed LDS [n][k]): lane = N column; lanes<16 K 0..7 & 16..23, lanes>=16 K 8..15 & 24..31
__device__ __forceinline__ void wmma_tile_step(const u16* sAh, const u16* sAl,
                                               const u16* sBh, const u16* sBl,
                                               int lane, int wm, int wn, f32x8 (&acc)[4][2]) {
  bf16x16 ah[4], al[4], bh[2], bl[2];
  const int lrow = lane & 15;
  const int aK = (lane >> 4) << 3;   // 0 or 8
  const int bK = (lane >> 4) << 3;   // 0 or 8
#pragma unroll
  for (int mt = 0; mt < 4; ++mt) {
    int r = wm * 64 + mt * 16 + lrow;
    uint4* p = reinterpret_cast<uint4*>(&ah[mt]);
    p[0] = *reinterpret_cast<const uint4*>(&sAh[r * 40 + aK]);
    p[1] = *reinterpret_cast<const uint4*>(&sAh[r * 40 + aK + 16]);
    uint4* q = reinterpret_cast<uint4*>(&al[mt]);
    q[0] = *reinterpret_cast<const uint4*>(&sAl[r * 40 + aK]);
    q[1] = *reinterpret_cast<const uint4*>(&sAl[r * 40 + aK + 16]);
  }
#pragma unroll
  for (int nt = 0; nt < 2; ++nt) {
    int cL = wn * 32 + nt * 16 + lrow;
    uint4* p = reinterpret_cast<uint4*>(&bh[nt]);
    p[0] = *reinterpret_cast<const uint4*>(&sBh[cL * 40 + bK]);
    p[1] = *reinterpret_cast<const uint4*>(&sBh[cL * 40 + bK + 16]);
    uint4* q = reinterpret_cast<uint4*>(&bl[nt]);
    q[0] = *reinterpret_cast<const uint4*>(&sBl[cL * 40 + bK]);
    q[1] = *reinterpret_cast<const uint4*>(&sBl[cL * 40 + bK + 16]);
  }
#pragma unroll
  for (int mt = 0; mt < 4; ++mt)
#pragma unroll
    for (int nt = 0; nt < 2; ++nt) {
      acc[mt][nt] = __builtin_amdgcn_wmma_f32_16x16x32_bf16(
          false, ah[mt], false, bh[nt], (short)0, acc[mt][nt], false, false);
      acc[mt][nt] = __builtin_amdgcn_wmma_f32_16x16x32_bf16(
          false, ah[mt], false, bl[nt], (short)0, acc[mt][nt], false, false);
      acc[mt][nt] = __builtin_amdgcn_wmma_f32_16x16x32_bf16(
          false, al[mt], false, bh[nt], (short)0, acc[mt][nt], false, false);
    }
}

// ---------------- split-bf16 WMMA GEMM, TDM double-buffered ----------------
// C tile 128x128 = A[B x K] * Wt[N x K]^T, bf16x3 split, f32 accumulate.
// MODE 0: relu(acc+bias[col])        -> split store H[:, col]
// MODE 1: relu(1-(acc+bias[col]))    -> split store H[:, 512+col]
// MODE 2: pre = w0*x + w1*relu(acc+bias[col]) -> f32 store
template <int MODE>
__global__ __launch_bounds__(256)
void gemm_split_bf16_kernel(const u16* __restrict__ Ahi, const u16* __restrict__ Alo, int lda,
                            const u16* __restrict__ Whi, const u16* __restrict__ Wlo, int ldb,
                            int K,
                            const float* __restrict__ bias,
                            const float* __restrict__ xsrc,
                            const float* __restrict__ wadd,
                            u16* __restrict__ OutHi, u16* __restrict__ OutLo, int ldo, int ocol,
                            float* __restrict__ OutF) {
  constexpr int NBUF = HAVE_TDM ? 2 : 1;
  __shared__ u16 sAhi[NBUF][128 * 40];
  __shared__ u16 sAlo[NBUF][128 * 40];
  __shared__ u16 sBhi[NBUF][128 * 40];
  __shared__ u16 sBlo[NBUF][128 * 40];

  const int t = threadIdx.x;
  const int lane = t & 31, wid = t >> 5;
  const int wm = wid & 1, wn = wid >> 1;       // 2 x 4 wave grid; wave tile 64(M) x 32(N)
  const int m0 = blockIdx.y * 128, n0 = blockIdx.x * 128;

  f32x8 acc[4][2];
#pragma unroll
  for (int i = 0; i < 4; ++i)
#pragma unroll
    for (int j = 0; j < 2; ++j) acc[i][j] = (f32x8)0.0f;

#if HAVE_TDM
  const unsigned strideA = (unsigned)(lda >> 1);  // dwords
  const unsigned strideB = (unsigned)(ldb >> 1);
  auto issue_set = [&](int kt, int buf) {
    tdm_load_tile(Ahi + (size_t)m0 * lda + kt, (unsigned)(size_t)&sAhi[buf][0], strideA);
    tdm_load_tile(Alo + (size_t)m0 * lda + kt, (unsigned)(size_t)&sAlo[buf][0], strideA);
    tdm_load_tile(Whi + (size_t)n0 * ldb + kt, (unsigned)(size_t)&sBhi[buf][0], strideB);
    tdm_load_tile(Wlo + (size_t)n0 * ldb + kt, (unsigned)(size_t)&sBlo[buf][0], strideB);
  };
  const int nk = K >> 5;
  if (t < 32) issue_set(0, 0);                  // prologue DMA
  for (int ki = 0; ki < nk; ++ki) {
    const int cur = ki & 1;
    if (t < 32) {
      if (ki + 1 < nk) {
        issue_set((ki + 1) << 5, cur ^ 1);      // async prefetch of next tile-set
        __builtin_amdgcn_s_wait_tensorcnt(4);   // in-order: current set complete
      } else {
        __builtin_amdgcn_s_wait_tensorcnt(0);
      }
    }
    __syncthreads();
    wmma_tile_step(sAhi[cur], sAlo[cur], sBhi[cur], sBlo[cur], lane, wm, wn, acc);
    __syncthreads();
  }
#else
  for (int kt = 0; kt < K; kt += 32) {
    for (int c = t; c < 512; c += 256) {
      int row = c >> 2, kc = (c & 3) << 3;
      size_t ga = (size_t)(m0 + row) * lda + kt + kc;
      size_t gb = (size_t)(n0 + row) * ldb + kt + kc;
      *reinterpret_cast<uint4*>(&sAhi[0][row * 40 + kc]) = *reinterpret_cast<const uint4*>(&Ahi[ga]);
      *reinterpret_cast<uint4*>(&sAlo[0][row * 40 + kc]) = *reinterpret_cast<const uint4*>(&Alo[ga]);
      *reinterpret_cast<uint4*>(&sBhi[0][row * 40 + kc]) = *reinterpret_cast<const uint4*>(&Whi[gb]);
      *reinterpret_cast<uint4*>(&sBlo[0][row * 40 + kc]) = *reinterpret_cast<const uint4*>(&Wlo[gb]);
    }
    __syncthreads();
    wmma_tile_step(sAhi[0], sAlo[0], sBhi[0], sBlo[0], lane, wm, wn, acc);
    __syncthreads();
  }
#endif

  // ---- epilogue: C layout: N = lane%16, M = vgpr + 8*(lane/16) ----
  const float w0 = (MODE == 2) ? wadd[0] : 0.f;
  const float w1 = (MODE == 2) ? wadd[1] : 0.f;
#pragma unroll
  for (int mt = 0; mt < 4; ++mt)
#pragma unroll
    for (int nt = 0; nt < 2; ++nt) {
      int col = n0 + wn * 32 + nt * 16 + (lane & 15);
      float bc = bias[col];
#pragma unroll
      for (int r = 0; r < 8; ++r) {
        int row = m0 + wm * 64 + mt * 16 + ((lane >> 4) << 3) + r;
        float v = acc[mt][nt][r];
        if (MODE == 0) {
          float h = fmaxf(v + bc, 0.f);
          u16 hh, hl; split2(h, hh, hl);
          size_t o = (size_t)row * ldo + ocol + col;
          OutHi[o] = hh; OutLo[o] = hl;
        } else if (MODE == 1) {
          float h = fmaxf(1.f - (v + bc), 0.f);
          u16 hh, hl; split2(h, hh, hl);
          size_t o = (size_t)row * ldo + ocol + col;
          OutHi[o] = hh; OutLo[o] = hl;
        } else {
          float e = fmaxf(v + bc, 0.f);
          size_t o = (size_t)row * Dd + col;
          OutF[o] = w0 * xsrc[o] + w1 * e;
        }
      }
    }
}

// ---------------- sparsemax (bisection + exact segment solve) ----------------
__global__ __launch_bounds__(256)
void sparsemax_kernel(const float* __restrict__ pre, float* __restrict__ out) {
  __shared__ float red[256];
  const int t = threadIdx.x;
  const size_t base = (size_t)blockIdx.x * Dd;
  float z0 = pre[base + t], z1 = pre[base + 256 + t];

  red[t] = fmaxf(z0, z1); __syncthreads();
  for (int off = 128; off > 0; off >>= 1) { if (t < off) red[t] = fmaxf(red[t], red[t + off]); __syncthreads(); }
  float zmax = red[0];

  float lo = zmax - 1.0f, hi = zmax;
  for (int it = 0; it < 32; ++it) {
    float mid = 0.5f * (lo + hi);
    float s = fmaxf(z0 - mid, 0.f) + fmaxf(z1 - mid, 0.f);
    __syncthreads();
    red[t] = s; __syncthreads();
    for (int off = 128; off > 0; off >>= 1) { if (t < off) red[t] += red[t + off]; __syncthreads(); }
    if (red[0] >= 1.0f) lo = mid; else hi = mid;
  }
  float cnt = (z0 > lo ? 1.f : 0.f) + (z1 > lo ? 1.f : 0.f);
  float sum = (z0 > lo ? z0 : 0.f) + (z1 > lo ? z1 : 0.f);
  __syncthreads();
  red[t] = cnt; __syncthreads();
  for (int off = 128; off > 0; off >>= 1) { if (t < off) red[t] += red[t + off]; __syncthreads(); }
  float kcnt = red[0];
  __syncthreads();
  red[t] = sum; __syncthreads();
  for (int off = 128; off > 0; off >>= 1) { if (t < off) red[t] += red[t + off]; __syncthreads(); }
  float S = red[0];
  float tau = (S - 1.0f) / fmaxf(kcnt, 1.0f);
  out[base + t]       = fmaxf(z0 - tau, 0.f);
  out[base + 256 + t] = fmaxf(z1 - tau, 0.f);
}

// ---------------- launch ----------------
extern "C" void kernel_launch(void* const* d_in, const int* in_sizes, int n_in,
                              void* d_out, int out_size, void* d_ws, size_t ws_size,
                              hipStream_t stream) {
  const float* x       = (const float*)d_in[0];
  const float* W1      = (const float*)d_in[1];
  const float* b1      = (const float*)d_in[2];
  const float* centers = (const float*)d_in[3];
  const float* coefs   = (const float*)d_in[4];
  const float* gamma   = (const float*)d_in[5];
  const float* beta    = (const float*)d_in[6];
  const float* mean    = (const float*)d_in[7];
  const float* var     = (const float*)d_in[8];
  const float* Wout    = (const float*)d_in[9];
  const float* bout    = (const float*)d_in[10];
  const float* wadd    = (const float*)d_in[11];

  char* ws = (char*)d_ws;
  size_t off = 0;
  auto alloc = [&](size_t bytes) -> void* {
    void* p = ws + off;
    off = (off + bytes + 255) & ~(size_t)255;
    return p;
  };
  const size_t BD = (size_t)Bsz * Dd;
  u16*   Xhi  = (u16*)alloc(BD * 2);
  u16*   Xlo  = (u16*)alloc(BD * 2);
  u16*   Phi  = (u16*)alloc(BD * 4);       // [B, 1024]
  u16*   Plo  = (u16*)alloc(BD * 4);
  u16*   Hhi  = (u16*)alloc(BD * 4);       // [B, 1024] = [h1 | h2]
  u16*   Hlo  = (u16*)alloc(BD * 4);
  float* preb = (float*)alloc(BD * 4);
  u16*   W1thi = (u16*)alloc((size_t)512 * 512 * 2);     // [N=512, K=512]
  u16*   W1tlo = (u16*)alloc((size_t)512 * 512 * 2);
  u16*   Wpthi = (u16*)alloc((size_t)512 * 1024 * 2);    // [N=512, K=1024]
  u16*   Wptlo = (u16*)alloc((size_t)512 * 1024 * 2);
  u16*   Wothi = (u16*)alloc((size_t)512 * 1024 * 2);    // [N=512, K=1024]
  u16*   Wotlo = (u16*)alloc((size_t)512 * 1024 * 2);
  float* cvec = (float*)alloc(512 * 4);

  prep_x_kernel<<<(Bsz * Dd) / 256, 256, 0, stream>>>(x, gamma, beta, mean, var, Xhi, Xlo, Phi, Plo);
  split_wt_kernel<<<(512 * 512) / 256, 256, 0, stream>>>(W1, W1thi, W1tlo, 512, 512);
  split_wt_kernel<<<(1024 * 512) / 256, 256, 0, stream>>>(Wout, Wothi, Wotlo, 1024, 512);
  prep_wp_kernel<<<(512 * 512) / 256, 256, 0, stream>>>(centers, coefs, Wpthi, Wptlo);
  prep_cvec_kernel<<<512, 256, 0, stream>>>(centers, coefs, cvec);

  dim3 ggrid(512 / 128, Bsz / 128);  // (4, 128)
  // h1 = relu(X*W1 + b1) -> H[:, 0:512]
  gemm_split_bf16_kernel<0><<<ggrid, 256, 0, stream>>>(
      Xhi, Xlo, 512, W1thi, W1tlo, 512, 512, b1, nullptr, nullptr, Hhi, Hlo, 1024, 0, nullptr);
  // h2 = relu(1 - (P*Wp + cvec)) -> H[:, 512:1024]
  gemm_split_bf16_kernel<1><<<ggrid, 256, 0, stream>>>(
      Phi, Plo, 1024, Wpthi, Wptlo, 1024, 1024, cvec, nullptr, nullptr, Hhi, Hlo, 1024, 512, nullptr);
  // pre = w0*x + w1*relu(H*Wout + bout)
  gemm_split_bf16_kernel<2><<<ggrid, 256, 0, stream>>>(
      Hhi, Hlo, 1024, Wothi, Wotlo, 1024, 1024, bout, x, wadd, nullptr, nullptr, 0, 0, preb);
  sparsemax_kernel<<<Bsz, 256, 0, stream>>>(preb, (float*)d_out);
}
